// AlibiAttention_40561671143490
// MI455X (gfx1250) — matrix operand order repaired and checked by reading, not compiled, not run on hardware
//
#include <hip/hip_runtime.h>
#include <hip/hip_bf16.h>

// ---------------------------------------------------------------------------
// Problem constants (from reference): B=8, S=1024, C=768, H=12, D=64
// ---------------------------------------------------------------------------
#define BB   8
#define SS   1024
#define CC   768
#define HH   12
#define DD   64
#define NQKV (3 * HH * DD)      // 2304
#define MM   (BB * SS)          // 8192
#define WIDTH_ 32
#define RADIUS_ 3

typedef __attribute__((ext_vector_type(16))) __bf16 v16bf;
typedef __attribute__((ext_vector_type(8)))  float  v8f;

union FragB16 {
    unsigned int u[8];
    v16bf        v;
};

__device__ __forceinline__ unsigned short f2bf(float f) {
    unsigned int u = __float_as_uint(f);
    unsigned int r = u + 0x7FFFu + ((u >> 16) & 1u);   // round-to-nearest-even
    return (unsigned short)(r >> 16);
}
__device__ __forceinline__ float bflo(unsigned int w) {
    return __uint_as_float(w << 16);
}
__device__ __forceinline__ float bfhi(unsigned int w) {
    return __uint_as_float(w & 0xFFFF0000u);
}

// ---------------------------------------------------------------------------
// CDNA5 async global->LDS DMA (ASYNCcnt-tracked, no VGPR round trip).
// Inline asm per bridge doc (portable across ROCm 7.2 / amdgpu-toolchain).
// ---------------------------------------------------------------------------
__device__ __forceinline__ void async_b128(const unsigned short* g, unsigned lds_off) {
    unsigned long long ga = (unsigned long long)(uintptr_t)g;
    asm volatile("global_load_async_to_lds_b128 %0, %1, off"
                 :: "v"(lds_off), "v"(ga) : "memory");
}
__device__ __forceinline__ void wait_async0() {
    asm volatile("s_wait_asynccnt 0" ::: "memory");
}

// ---------------------------------------------------------------------------
// fp32 -> bf16 elementwise (2 elements / thread, packed dword stores)
// ---------------------------------------------------------------------------
__global__ __launch_bounds__(256) void cvt_f32_bf16(const float* __restrict__ in,
                                                    unsigned short* __restrict__ out,
                                                    int n2) {
    int i = blockIdx.x * 256 + threadIdx.x;
    if (i < n2) {
        float2 f = ((const float2*)in)[i];
        unsigned int p = (unsigned int)f2bf(f.x) | ((unsigned int)f2bf(f.y) << 16);
        ((unsigned int*)out)[i] = p;
    }
}

// ---------------------------------------------------------------------------
// bf16 WMMA GEMM:  C[M,N] = A[M,K] * B[N,K]^T  (+ bias)
// A, B row-major bf16.  Workgroup = 256 thr (8 waves), tile 128(M) x 64(N).
// Wave grid 4x2, each wave computes a 32x32 C tile (2x2 16x16 fragments).
// LDS tiles padded to 40 halves/row (80B pitch: 16B aligned, conflict-free
// fragment rows). Double-buffered, filled with global_load_async_to_lds_b128.
// ---------------------------------------------------------------------------
#define LDS_STRIDE 40

__device__ __forceinline__ void loadFragA(const unsigned short* As, int rbase,
                                          int lane, FragB16& f) {
    // 16-bit A 16x32 layout: lanes 0-15 -> M=lane; VGPR v holds K pair
    // k = (v>>2)*16 + (lane>>4)*8 + (v&3)*2
    int m  = lane & 15;
    int kh = lane >> 4;
    const unsigned short* p = As + (rbase + m) * LDS_STRIDE;
#pragma unroll
    for (int v = 0; v < 8; ++v) {
        int k = ((v >> 2) << 4) + (kh << 3) + ((v & 3) << 1);
        f.u[v] = *(const unsigned int*)(p + k);
    }
}

__device__ __forceinline__ void loadFragB(const unsigned short* Bs, int cbase,
                                          int lane, FragB16& f) {
    // 16-bit B 32x16 layout: N = lane&15; VGPR v holds K pair
    // k = (v>>2)*16 + (lane>>4)*8 + (v&3)*2
    int n  = lane & 15;
    int kh = lane >> 4;
    const unsigned short* p = Bs + (cbase + n) * LDS_STRIDE + (kh << 3);
#pragma unroll
    for (int v = 0; v < 8; ++v) {
        f.u[v] = *(const unsigned int*)(p + ((v >> 2) << 4) + ((v & 3) << 1));
    }
}

template <int OUT_BF16, int HAS_BIAS>
__global__ __launch_bounds__(256) void gemm_bf16_wmma(
    const unsigned short* __restrict__ A,   // [M,K] bf16
    const unsigned short* __restrict__ Bm,  // [N,K] bf16
    void* __restrict__ Cout,                // [M,N] f32 or bf16
    const float* __restrict__ bias,         // [N] or null
    int M, int N, int K) {
    __shared__ unsigned short As[2][128 * LDS_STRIDE];
    __shared__ unsigned short Bs[2][64 * LDS_STRIDE];

    const int t       = threadIdx.x;
    const int lane    = t & 31;
    const int wave    = t >> 5;
    const int wm      = wave >> 1;      // 0..3 -> M sub-tile
    const int wn      = wave & 1;       // 0..1 -> N sub-tile
    const int colBase = blockIdx.x * 64;
    const int rowBase = blockIdx.y * 128;

    // Global->LDS mapping: A tile 128 rows x 64B (2 lanes/row, 32B each),
    //                      B tile  64 rows x 64B (4 lanes/row, 16B each)
    const int arow = t >> 1, aseg = t & 1;
    const int brow = t >> 2, bseg = t & 3;

    const unsigned short* gA = A + (rowBase + arow) * K + aseg * 16;
    const unsigned short* gB = Bm + (colBase + brow) * K + bseg * 8;

    v8f acc[2][2];
#pragma unroll
    for (int i = 0; i < 2; ++i)
#pragma unroll
        for (int j = 0; j < 2; ++j)
#pragma unroll
            for (int e = 0; e < 8; ++e) acc[i][j][e] = 0.0f;

    // ---- prologue: stage tile 0 into buffer 0 ----
    {
        unsigned la = (unsigned)(uintptr_t)(&As[0][0] + arow * LDS_STRIDE + aseg * 16);
        async_b128(gA, la);
        async_b128(gA + 8, la + 16);
        unsigned lb = (unsigned)(uintptr_t)(&Bs[0][0] + brow * LDS_STRIDE + bseg * 8);
        async_b128(gB, lb);
    }
    wait_async0();
    __syncthreads();

    const int nk = K >> 5;
    for (int it = 0; it < nk; ++it) {
        const int cur = it & 1;
        const int nxt = cur ^ 1;
        const bool more = (it + 1) < nk;

        // ---- issue async DMA for next tile into the other buffer ----
        if (more) {
            int k1 = (it + 1) << 5;
            unsigned la = (unsigned)(uintptr_t)(&As[nxt][0] + arow * LDS_STRIDE + aseg * 16);
            async_b128(gA + k1, la);
            async_b128(gA + k1 + 8, la + 16);
            unsigned lb = (unsigned)(uintptr_t)(&Bs[nxt][0] + brow * LDS_STRIDE + bseg * 8);
            async_b128(gB + k1, lb);
        }

        // ---- compute current tile: 2x2 fragments per wave ----
        FragB16 a0, a1, b0, b1;
        loadFragA(&As[cur][0], wm * 32, lane, a0);
        loadFragA(&As[cur][0], wm * 32 + 16, lane, a1);
        loadFragB(&Bs[cur][0], wn * 32, lane, b0);
        loadFragB(&Bs[cur][0], wn * 32 + 16, lane, b1);

        acc[0][0] = __builtin_amdgcn_wmma_f32_16x16x32_bf16(
            false, a0.v, false, b0.v, (short)0, acc[0][0], false, false);
        acc[0][1] = __builtin_amdgcn_wmma_f32_16x16x32_bf16(
            false, a0.v, false, b1.v, (short)0, acc[0][1], false, false);
        acc[1][0] = __builtin_amdgcn_wmma_f32_16x16x32_bf16(
            false, a1.v, false, b0.v, (short)0, acc[1][0], false, false);
        acc[1][1] = __builtin_amdgcn_wmma_f32_16x16x32_bf16(
            false, a1.v, false, b1.v, (short)0, acc[1][1], false, false);

        // next buffer must be fully written before anyone reads it
        if (more) wait_async0();
        __syncthreads();
    }

    // ---- store: f32 C/D layout: lane&15 = N, (lane>>4)*8+v = M ----
    const int cn = lane & 15;
    const int rh = lane >> 4;
#pragma unroll
    for (int fi = 0; fi < 2; ++fi) {
#pragma unroll
        for (int fj = 0; fj < 2; ++fj) {
            int col  = colBase + wn * 32 + fj * 16 + cn;
            float bv = HAS_BIAS ? bias[col] : 0.0f;
#pragma unroll
            for (int v = 0; v < 8; ++v) {
                int row   = rowBase + wm * 32 + fi * 16 + rh * 8 + v;
                float val = acc[fi][fj][v] + bv;
                if (OUT_BF16)
                    ((unsigned short*)Cout)[(size_t)row * N + col] = f2bf(val);
                else
                    ((float*)Cout)[(size_t)row * N + col] = val;
            }
        }
    }
}

// ---------------------------------------------------------------------------
// Windowed ALiBi attention, online softmax, one thread per (b, h, q).
// qkv: bf16 [B, S, 3, H, D]  ->  attout: bf16 [B, S, H*D]  (transpose folded)
// ---------------------------------------------------------------------------
__global__ __launch_bounds__(256) void attn_window(
    const unsigned short* __restrict__ qkv,
    unsigned short* __restrict__ attout) {
    int tid = blockIdx.x * 256 + threadIdx.x;
    if (tid >= BB * HH * SS) return;
    int q   = tid % SS;
    int rem = tid / SS;
    int h   = rem % HH;
    int b   = rem / HH;

    const int r = q >> 5;       // row in 32x32 grid
    const int c = q & 31;       // col
    const float slope   = 1.0f / (float)(1 << ((h + 2) >> 1));
    const int   use_col = h & 1;

    const int rowOff = (b * SS + q) * NQKV + h * DD;
    const unsigned int* qp = (const unsigned int*)(qkv + rowOff);

    float qf[DD];
#pragma unroll
    for (int i = 0; i < DD / 2; ++i) {
        unsigned int w = qp[i];
        qf[2 * i]     = bflo(w);
        qf[2 * i + 1] = bfhi(w);
    }

    float mrun = -1.0e30f, lrun = 0.0f;
    float oacc[DD];
#pragma unroll
    for (int i = 0; i < DD; ++i) oacc[i] = 0.0f;

    for (int dr = -RADIUS_; dr <= RADIUS_; ++dr) {
        int rr = r + dr;
        if (rr < 0 || rr >= WIDTH_) continue;
        for (int dc = -RADIUS_; dc <= RADIUS_; ++dc) {
            int cc = c + dc;
            if (cc < 0 || cc >= WIDTH_) continue;
            int kpos  = (rr << 5) | cc;
            int kbase = (b * SS + kpos) * NQKV + h * DD;

            const unsigned int* kp = (const unsigned int*)(qkv + kbase + HH * DD);
            float s = 0.0f;
#pragma unroll
            for (int i = 0; i < DD / 2; ++i) {
                unsigned int w = kp[i];
                s += qf[2 * i] * bflo(w) + qf[2 * i + 1] * bfhi(w);
            }
            float delta = use_col ? (float)dc : (float)dr;
            s = s * 0.125f - slope * fabsf(delta);    // SCALE = 64^-0.5

            float nm = fmaxf(mrun, s);
            float ef = __expf(mrun - nm);
            float es = __expf(s - nm);
            lrun = lrun * ef + es;

            const unsigned int* vp = (const unsigned int*)(qkv + kbase + 2 * HH * DD);
#pragma unroll
            for (int i = 0; i < DD / 2; ++i) {
                unsigned int w = vp[i];
                oacc[2 * i]     = oacc[2 * i] * ef + es * bflo(w);
                oacc[2 * i + 1] = oacc[2 * i + 1] * ef + es * bfhi(w);
            }
            mrun = nm;
        }
    }

    float inv = 1.0f / lrun;
    unsigned int* op = (unsigned int*)(attout + (b * SS + q) * CC + h * DD);
#pragma unroll
    for (int i = 0; i < DD / 2; ++i) {
        op[i] = (unsigned int)f2bf(oacc[2 * i] * inv) |
                ((unsigned int)f2bf(oacc[2 * i + 1] * inv) << 16);
    }
}

// ---------------------------------------------------------------------------
// Host launch
// ---------------------------------------------------------------------------
extern "C" void kernel_launch(void* const* d_in, const int* in_sizes, int n_in,
                              void* d_out, int out_size, void* d_ws, size_t ws_size,
                              hipStream_t stream) {
    const float* x      = (const float*)d_in[0];   // [B,S,C]
    const float* w_qkv  = (const float*)d_in[1];   // [2304,768]
    const float* w_proj = (const float*)d_in[2];   // [768,768]
    const float* b_proj = (const float*)d_in[3];   // [768]

    char* ws = (char*)d_ws;
    // workspace layout (bf16 buffers)
    unsigned short* xh     = (unsigned short*)(ws);                    // 12.6 MB
    unsigned short* wqkvh  = (unsigned short*)(ws + 12582912);         //  3.5 MB
    unsigned short* wprojh = (unsigned short*)(ws + 16121856);         //  1.2 MB
    unsigned short* qkvh   = (unsigned short*)(ws + 17301504);         // 37.7 MB
    unsigned short* atth   = (unsigned short*)(ws + 55050240);         // 12.6 MB

    // 1. downconvert inputs to bf16
    {
        int n2 = (MM * CC) / 2;
        cvt_f32_bf16<<<(n2 + 255) / 256, 256, 0, stream>>>(x, xh, n2);
        n2 = (NQKV * CC) / 2;
        cvt_f32_bf16<<<(n2 + 255) / 256, 256, 0, stream>>>(w_qkv, wqkvh, n2);
        n2 = (CC * CC) / 2;
        cvt_f32_bf16<<<(n2 + 255) / 256, 256, 0, stream>>>(w_proj, wprojh, n2);
    }

    // 2. qkv = x @ w_qkv^T   (M=8192, N=2304, K=768), bf16 out
    gemm_bf16_wmma<1, 0><<<dim3(NQKV / 64, MM / 128), 256, 0, stream>>>(
        xh, wqkvh, (void*)qkvh, nullptr, MM, NQKV, CC);

    // 3. windowed ALiBi attention -> bf16 [B,S,C]
    attn_window<<<(BB * HH * SS) / 256, 256, 0, stream>>>(qkvh, atth);

    // 4. out = att @ w_proj^T + b   (M=8192, N=768, K=768), f32 out
    gemm_bf16_wmma<0, 1><<<dim3(CC / 64, MM / 128), 256, 0, stream>>>(
        atth, wprojh, d_out, b_proj, MM, CC, CC);
}
